// get_model_21114059227689
// MI455X (gfx1250) — matrix operand order repaired and checked by reading, not compiled, not run on hardware
//
#include <hip/hip_runtime.h>

typedef _Float16 v16h __attribute__((ext_vector_type(16)));
typedef _Float16 v8h  __attribute__((ext_vector_type(8)));
typedef float    v8f  __attribute__((ext_vector_type(8)));

namespace {

constexpr int   kB     = 2;
constexpr int   kN     = 16384;
constexpr float kBnInv = 0.9999950000375f;   // 1/sqrt(1+1e-5)

// ---------------- data-movement / irregular kernels ----------------

// xyz [B,6,N] -> l0_points [B*N,6], l0_xyz [B*N,3], and output T(l0_xyz) [B,3,N]
__global__ void k_split(const float* __restrict__ xyz, float* __restrict__ l0p,
                        float* __restrict__ l0x, float* __restrict__ out2,
                        int N, int total) {
  int t = blockIdx.x * blockDim.x + threadIdx.x;
  if (t >= total) return;
  int n  = t % N;
  int ch = (t / N) % 6;
  int b  = t / (6 * N);
  float v = xyz[t];
  l0p[((size_t)b * N + n) * 6 + ch] = v;
  if (ch < 3) {
    l0x[((size_t)b * N + n) * 3 + ch] = v;
    out2[((size_t)b * 3 + ch) * N + n] = v;
  }
}

// farthest point sampling; one 256-thread block per batch
__global__ void k_fps(const float* __restrict__ xyz, int n, int npoint,
                      int* __restrict__ idx, float* __restrict__ dist) {
  const int b = blockIdx.x, t = threadIdx.x, T = 256;
  const float* X = xyz + (size_t)b * n * 3;
  float* D = dist + (size_t)b * n;
  int*   I = idx  + (size_t)b * npoint;
  for (int i = t; i < n; i += T) D[i] = 1e10f;
  __shared__ float sval[256];
  __shared__ int   sidx[256];
  __shared__ int   cur;
  if (t == 0) { cur = 0; I[0] = 0; }
  __syncthreads();
  for (int it = 1; it < npoint; ++it) {
    int c = cur;
    float cx = X[c*3+0], cy = X[c*3+1], cz = X[c*3+2];
    float best = -1.0f; int bi = 0;
    for (int i = t; i < n; i += T) {
      float dx = X[i*3+0]-cx, dy = X[i*3+1]-cy, dz = X[i*3+2]-cz;
      float d = dx*dx + dy*dy + dz*dz;
      float m = fminf(D[i], d);
      D[i] = m;
      if (m > best) { best = m; bi = i; }
    }
    sval[t] = best; sidx[t] = bi;
    __syncthreads();
    for (int s = 128; s > 0; s >>= 1) {
      if (t < s && sval[t+s] > sval[t]) { sval[t] = sval[t+s]; sidx[t] = sidx[t+s]; }
      __syncthreads();
    }
    if (t == 0) { cur = sidx[0]; I[it] = sidx[0]; }
    __syncthreads();
  }
}

// new_xyz[bm,d] = xyz[b, fidx[bm], d]
__global__ void k_gather3(const float* __restrict__ xyz, const int* __restrict__ fidx,
                          int n, int M, float* __restrict__ out, int total) {
  int t = blockIdx.x * blockDim.x + threadIdx.x;
  if (t >= total) return;
  int d = t % 3, bm = t / 3, b = bm / M;
  int g = fidx[bm];
  out[t] = xyz[((size_t)b * n + g) * 3 + d];
}

// ball query: first K indices (ascending) with sqdist <= r2, padded with first hit
__global__ void k_ball(const float* __restrict__ xyz, const float* __restrict__ newxyz,
                       int n, int M, float r2, int K, int* __restrict__ gidx, int total) {
  int t = blockIdx.x * blockDim.x + threadIdx.x;
  if (t >= total) return;
  int b = t / M;
  const float* X = xyz + (size_t)b * n * 3;
  float qx = newxyz[t*3+0], qy = newxyz[t*3+1], qz = newxyz[t*3+2];
  int* out = gidx + (size_t)t * K;
  int cnt = 0, first = 0;
  bool found = false;
  for (int i = 0; i < n && cnt < K; ++i) {
    float dx = X[i*3+0]-qx, dy = X[i*3+1]-qy, dz = X[i*3+2]-qz;
    float d = dx*dx + dy*dy + dz*dz;
    if (d <= r2) {
      if (!found) { first = i; found = true; }
      out[cnt++] = i;
    }
  }
  for (; cnt < K; ++cnt) out[cnt] = first;
}

// grouped features: rows = B*M*K, cols = C+3  ([points | xyz - new_xyz])
__global__ void k_group(const float* __restrict__ pts, const float* __restrict__ xyz,
                        const float* __restrict__ newxyz, const int* __restrict__ gidx,
                        int n, int M, int K, int C, float* __restrict__ G, int total) {
  int t = blockIdx.x * blockDim.x + threadIdx.x;
  if (t >= total) return;
  int Cg  = C + 3;
  int col = t % Cg;
  int row = t / Cg;            // in [0, B*M*K)
  int g   = gidx[row];
  int bm  = row / K;
  int b   = bm / M;
  if (col < C) {
    G[t] = pts[((size_t)b * n + g) * C + col];
  } else {
    int d = col - C;
    G[t] = xyz[((size_t)b * n + g) * 3 + d] - newxyz[(size_t)bm * 3 + d];
  }
}

// max over group dim K; write at column offset into concat buffer
__global__ void k_maxpool(const float* __restrict__ Xin, float* __restrict__ Y,
                          int M, int K, int C, int ldY, int colOff, int total) {
  int t = blockIdx.x * blockDim.x + threadIdx.x;
  if (t >= total) return;
  int c = t % C, bm = t / C;
  float v = -3.4e38f;
  const float* src = Xin + (size_t)bm * K * C + c;
  for (int k = 0; k < K; ++k) v = fmaxf(v, src[(size_t)k * C]);
  Y[(size_t)bm * ldY + colOff + c] = v;
}

// 3 nearest neighbors of xyz1[t] among xyz2 (per batch), inverse-distance weights
__global__ void k_knn3(const float* __restrict__ xyz1, const float* __restrict__ xyz2,
                       int n1, int n2, int* __restrict__ idx3, float* __restrict__ w3, int total) {
  int t = blockIdx.x * blockDim.x + threadIdx.x;
  if (t >= total) return;
  int b = t / n1;
  float qx = xyz1[t*3+0], qy = xyz1[t*3+1], qz = xyz1[t*3+2];
  const float* X2 = xyz2 + (size_t)b * n2 * 3;
  float d0 = 3.4e38f, d1 = 3.4e38f, d2 = 3.4e38f;
  int   i0 = 0, i1 = 0, i2 = 0;
  for (int i = 0; i < n2; ++i) {
    float dx = X2[i*3+0]-qx, dy = X2[i*3+1]-qy, dz = X2[i*3+2]-qz;
    float d = dx*dx + dy*dy + dz*dz;
    if (d < d0)      { d2=d1; i2=i1; d1=d0; i1=i0; d0=d; i0=i; }
    else if (d < d1) { d2=d1; i2=i1; d1=d; i1=i; }
    else if (d < d2) { d2=d; i2=i; }
  }
  float w0 = 1.f/(d0+1e-8f), w1 = 1.f/(d1+1e-8f), w2 = 1.f/(d2+1e-8f);
  float s = w0 + w1 + w2;
  idx3[t*3+0] = i0; idx3[t*3+1] = i1; idx3[t*3+2] = i2;
  w3[t*3+0] = w0/s; w3[t*3+1] = w1/s; w3[t*3+2] = w2/s;
}

// interpolate pts2 with 3-NN weights into X at column offset
__global__ void k_interp(const float* __restrict__ pts2, const int* __restrict__ idx3,
                         const float* __restrict__ w3, float* __restrict__ X,
                         int n1, int n2, int C2, int ld, int colOff, int total) {
  int t = blockIdx.x * blockDim.x + threadIdx.x;
  if (t >= total) return;
  int c = t % C2, row = t / C2, b = row / n1;
  const float* P = pts2 + (size_t)b * n2 * C2;
  float v = 0.f;
  for (int j = 0; j < 3; ++j)
    v += w3[row*3+j] * P[(size_t)idx3[row*3+j] * C2 + c];
  X[(size_t)row * ld + colOff + c] = v;
}

__global__ void k_copy2d(const float* __restrict__ src, float* __restrict__ dst,
                         int C, int ld, int colOff, int total) {
  int t = blockIdx.x * blockDim.x + threadIdx.x;
  if (t >= total) return;
  int c = t % C, r = t / C;
  dst[(size_t)r * ld + colOff + c] = src[t];
}

// [B*M, C] row-major -> [B, C, M]
__global__ void k_tout(const float* __restrict__ src, float* __restrict__ dst,
                       int Mpts, int C, int total) {
  int t = blockIdx.x * blockDim.x + threadIdx.x;
  if (t >= total) return;
  int c = t % C, row = t / C, m = row % Mpts, b = row / Mpts;
  dst[((size_t)b * C + c) * Mpts + m] = src[t];
}

// f32 [rows,C] -> f16 [*, CP] zero-padded (rows padded via total)
__global__ void k_cvt(const float* __restrict__ src, _Float16* __restrict__ dst,
                      int rows, int C, int CP, int total) {
  int t = blockIdx.x * blockDim.x + threadIdx.x;
  if (t >= total) return;
  int c = t % CP, r = t / CP;
  dst[t] = (r < rows && c < C) ? (_Float16)src[(size_t)r * C + c] : (_Float16)0.f;
}

// ---------------- WMMA GEMM: Y = act((X·Wᵀ + b) * g/√(1+ε) + β) ----------------
// Xh: rowsP x CinP f16 (zero padded), Wh: CoutP x CinP f16.
// One wave computes a 16x16 tile with v_wmma_f32_16x16x32_f16; 4 waves/block.
// A tile per lane: rows = rowTile + (lane&15); K chunk = two contiguous 8-half runs
// at k + half*8 and k + 16 + half*8 (ISA 16-bit A 16x32 layout).
// B tile per lane: col = colTile + (lane&15); K chunk = two contiguous 8-half runs
// at k + half*8 and k + 16 + half*8 (matches A).  EXEC is all-ones through the K loop.
__global__ void __launch_bounds__(128)
k_gemm(const _Float16* __restrict__ Xh, const _Float16* __restrict__ Wh,
       const float* __restrict__ bias, const float* __restrict__ gamma,
       const float* __restrict__ beta, float* __restrict__ Y,
       int rows, int Cout, int CinP, int CoutP, int ldY, int colOff, int mode) {
  const int lane    = threadIdx.x & 31;
  const int wave    = threadIdx.x >> 5;
  const int rowTile = blockIdx.x * 16;
  const int colTile = (blockIdx.y * 4 + wave) * 16;
  if (colTile >= CoutP) return;                    // wave-uniform
  const int half = lane >> 4, l15 = lane & 15;

  const _Float16* xa = Xh + (size_t)(rowTile + l15) * CinP;
  const _Float16* wb = Wh + (size_t)(colTile + l15) * CinP;

  v8f acc = {};
  for (int k = 0; k < CinP; k += 32) {
    union { v16h v; v8h h[2]; } a, b;
    a.h[0] = *(const v8h*)(xa + k + half * 8);
    a.h[1] = *(const v8h*)(xa + k + 16 + half * 8);
    b.h[0] = *(const v8h*)(wb + k + half * 8);
    b.h[1] = *(const v8h*)(wb + k + 16 + half * 8);
    acc = __builtin_amdgcn_wmma_f32_16x16x32_f16(
        /*neg_a=*/false, a.v, /*neg_b=*/false, b.v,
        /*c_mod=*/(short)0, acc, /*reuse_a=*/false, /*reuse_b=*/false);
  }

  int bcol = colTile + l15;
  if (bcol >= Cout) return;                        // store-only divergence
  float bb = bias[bcol];
  float g = 1.f, bt = 0.f;
  if (mode) { g = gamma[bcol] * kBnInv; bt = beta[bcol]; }
#pragma unroll
  for (int r = 0; r < 8; ++r) {
    int orow = rowTile + half * 8 + r;             // C/D layout: lanes16-31 hold M+8
    if (orow < rows) {
      float v = acc[r] + bb;
      if (mode) { v = fmaf(v, g, bt); v = v > 0.f ? v : 0.f; }
      Y[(size_t)orow * ldY + colOff + bcol] = v;
    }
  }
}

// ---------------- host-side orchestration ----------------

struct Lay { const float *w, *b, *g, *bt; int cin, cout; };

struct Bump {
  char* base; size_t off;
  void* take(size_t bytes) {
    off = (off + 255) & ~(size_t)255;
    void* r = base + off;
    off += bytes;
    return r;
  }
};

inline int ceil_div(int a, int b) { return (a + b - 1) / b; }

static void run_gemm(hipStream_t s, Bump& ws, const float* X, int rows, const Lay& L,
                     float* Y, int ldY, int colOff, int mode) {
  size_t mark = ws.off;
  int rowsP = (rows + 15) & ~15;
  int CinP  = (L.cin + 31) & ~31;
  int CoutP = (L.cout + 15) & ~15;
  _Float16* Xh = (_Float16*)ws.take((size_t)rowsP * CinP * sizeof(_Float16));
  _Float16* Wh = (_Float16*)ws.take((size_t)CoutP * CinP * sizeof(_Float16));
  int tx = rowsP * CinP;
  k_cvt<<<ceil_div(tx, 256), 256, 0, s>>>(X, Xh, rows, L.cin, CinP, tx);
  int tw = CoutP * CinP;
  k_cvt<<<ceil_div(tw, 256), 256, 0, s>>>(L.w, Wh, L.cout, L.cin, CinP, tw);
  dim3 grid(rowsP / 16, ceil_div(CoutP, 64));
  k_gemm<<<grid, 128, 0, s>>>(Xh, Wh, L.b, L.g, L.bt, Y, rows, L.cout, CinP, CoutP,
                              ldY, colOff, mode);
  ws.off = mark;
}

// one set-abstraction MSG stage
static void run_sa(hipStream_t s, Bump& ws, const float* xyz_in, const float* pts_in,
                   int n, int C, int M, const float* r2, const int* Ks,
                   const Lay* br0, const Lay* br1, float** out_xyz, float** out_pts) {
  float* newxyz = (float*)ws.take((size_t)kB * M * 3 * sizeof(float));
  int ctot = br0[1].cout + br1[1].cout;
  float* newpts = (float*)ws.take((size_t)kB * M * ctot * sizeof(float));
  size_t mark = ws.off;

  float* dist = (float*)ws.take((size_t)kB * n * sizeof(float));
  int*   fidx = (int*)ws.take((size_t)kB * M * sizeof(int));
  k_fps<<<kB, 256, 0, s>>>(xyz_in, n, M, fidx, dist);
  k_gather3<<<ceil_div(kB * M * 3, 256), 256, 0, s>>>(xyz_in, fidx, n, M, newxyz, kB * M * 3);

  const Lay* brs[2] = { br0, br1 };
  int colOff = 0;
  for (int j = 0; j < 2; ++j) {
    size_t m2 = ws.off;
    int K = Ks[j];
    int rowsG = kB * M * K;
    int Cg = C + 3;
    int* gidx = (int*)ws.take((size_t)rowsG * sizeof(int));
    k_ball<<<ceil_div(kB * M, 256), 256, 0, s>>>(xyz_in, newxyz, n, M, r2[j], K, gidx, kB * M);
    float* G = (float*)ws.take((size_t)rowsG * Cg * sizeof(float));
    k_group<<<ceil_div(rowsG * Cg, 256), 256, 0, s>>>(pts_in, xyz_in, newxyz, gidx,
                                                      n, M, K, C, G, rowsG * Cg);
    const Lay* L = brs[j];
    float* H0 = (float*)ws.take((size_t)rowsG * L[0].cout * sizeof(float));
    run_gemm(s, ws, G, rowsG, L[0], H0, L[0].cout, 0, 1);
    float* H1 = (float*)ws.take((size_t)rowsG * L[1].cout * sizeof(float));
    run_gemm(s, ws, H0, rowsG, L[1], H1, L[1].cout, 0, 1);
    k_maxpool<<<ceil_div(kB * M * L[1].cout, 256), 256, 0, s>>>(H1, newpts, M, K, L[1].cout,
                                                                ctot, colOff, kB * M * L[1].cout);
    colOff += L[1].cout;
    ws.off = m2;
  }
  ws.off = mark;
  *out_xyz = newxyz;
  *out_pts = newpts;
}

// one feature-propagation stage
static float* run_fp(hipStream_t s, Bump& ws, const float* xyz1, const float* xyz2,
                     int n1, int n2, const float* pts1, int C1,
                     const float* pts2, int C2, const Lay* L) {
  float* out = (float*)ws.take((size_t)kB * n1 * L[1].cout * sizeof(float));
  size_t mark = ws.off;
  int rows = kB * n1;
  int* idx3 = (int*)ws.take((size_t)rows * 3 * sizeof(int));
  float* w3 = (float*)ws.take((size_t)rows * 3 * sizeof(float));
  k_knn3<<<ceil_div(rows, 256), 256, 0, s>>>(xyz1, xyz2, n1, n2, idx3, w3, rows);
  int Cx = C1 + C2;
  float* X = (float*)ws.take((size_t)rows * Cx * sizeof(float));
  k_copy2d<<<ceil_div(rows * C1, 256), 256, 0, s>>>(pts1, X, C1, Cx, 0, rows * C1);
  k_interp<<<ceil_div(rows * C2, 256), 256, 0, s>>>(pts2, idx3, w3, X, n1, n2, C2, Cx, C1,
                                                    rows * C2);
  float* H0 = (float*)ws.take((size_t)rows * L[0].cout * sizeof(float));
  run_gemm(s, ws, X, rows, L[0], H0, L[0].cout, 0, 1);
  run_gemm(s, ws, H0, rows, L[1], out, L[1].cout, 0, 1);
  ws.off = mark;
  return out;
}

}  // namespace

extern "C" void kernel_launch(void* const* d_in, const int* in_sizes, int n_in,
                              void* d_out, int out_size, void* d_ws, size_t ws_size,
                              hipStream_t stream) {
  const float* xyz_in = (const float*)d_in[0];
  float* out = (float*)d_out;
  Bump ws{ (char*)d_ws, 0 };

  // ---- parse params (jax pytree order: dict keys sorted; block keys b,bt,g,w) ----
  int cur = 1;
  auto nextp = [&]() { return (const float*)d_in[cur++]; };
  auto block = [&](int cin, int cout) {
    Lay L; L.b = nextp(); L.bt = nextp(); L.g = nextp(); L.w = nextp();
    L.cin = cin; L.cout = cout; return L;
  };
  auto convp = [&](int cin, int cout) {
    Lay L; L.b = nextp(); L.w = nextp(); L.g = nullptr; L.bt = nullptr;
    L.cin = cin; L.cout = cout; return L;
  };

  Lay conv1  = block(32, 16);
  Lay dist1  = block(512, 256);
  Lay dist2  = convp(256, 1);
  Lay fp1L[2] = { block(134, 64),  block(64, 32)   };
  Lay fp2L[2] = { block(320, 128), block(128, 128) };
  Lay fp3L[2] = { block(768, 256), block(256, 256) };
  Lay off1   = block(512, 256);
  Lay off2   = convp(256, 3);
  Lay sa1b0[2] = { block(9, 32),   block(32, 32)   };
  Lay sa1b1[2] = { block(9, 32),   block(32, 32)   };
  Lay sa2b0[2] = { block(67, 64),  block(64, 128)  };
  Lay sa2b1[2] = { block(67, 64),  block(64, 128)  };
  Lay sa3b0[2] = { block(259, 196), block(196, 256) };
  Lay sa3b1[2] = { block(259, 196), block(196, 256) };

  // ---- output section offsets ----
  float* out0 = out;                       // T(x)         [B,16,N]
  float* out1 = out + 524288;              // T(l3_points) [B,512,256]
  float* out2 = out + 786432;              // T(l0_xyz)    [B,3,N]
  float* out3 = out + 884736;              // T(l3_xyz)    [B,3,256]
  float* out4 = out + 886272;              // T(off)       [B,3,256]
  float* out5 = out + 887808;              // T(dist)      [B,1,256]

  // ---- stage 0: input split / transpose ----
  float* l0p = (float*)ws.take((size_t)kB * kN * 6 * sizeof(float));
  float* l0x = (float*)ws.take((size_t)kB * kN * 3 * sizeof(float));
  k_split<<<ceil_div(kB * 6 * kN, 256), 256, 0, stream>>>(xyz_in, l0p, l0x, out2,
                                                          kN, kB * 6 * kN);

  // ---- SA stages ----
  const float r2_1[2] = { 0.0025f * 0.0025f, 0.005f * 0.005f };
  const float r2_2[2] = { 0.005f * 0.005f,   0.01f * 0.01f   };
  const float r2_3[2] = { 0.01f * 0.01f,     0.02f * 0.02f   };
  const int   Ks[2]   = { 16, 32 };

  float *l1x, *l1p, *l2x, *l2p, *l3x, *l3p;
  run_sa(stream, ws, l0x, l0p, kN,   6,   1024, r2_1, Ks, sa1b0, sa1b1, &l1x, &l1p);
  run_sa(stream, ws, l1x, l1p, 1024, 64,  512,  r2_2, Ks, sa2b0, sa2b1, &l2x, &l2p);
  run_sa(stream, ws, l2x, l2p, 512,  256, 256,  r2_3, Ks, sa3b0, sa3b1, &l3x, &l3p);

  // ---- FP stages ----
  float* l2pn = run_fp(stream, ws, l2x, l3x, 512,  256,  l2p, 256, l3p,  512, fp3L);
  float* l1pn = run_fp(stream, ws, l1x, l2x, 1024, 512,  l1p, 64,  l2pn, 256, fp2L);
  float* l0pn = run_fp(stream, ws, l0x, l1x, kN,   1024, l0p, 6,   l1pn, 128, fp1L);

  // ---- heads ----
  {
    size_t mark = ws.off;
    float* xfin = (float*)ws.take((size_t)kB * kN * 16 * sizeof(float));
    run_gemm(stream, ws, l0pn, kB * kN, conv1, xfin, 16, 0, 1);
    k_tout<<<ceil_div(kB * kN * 16, 256), 256, 0, stream>>>(xfin, out0, kN, 16, kB * kN * 16);
    ws.off = mark;
  }
  k_tout<<<ceil_div(kB * 256 * 512, 256), 256, 0, stream>>>(l3p, out1, 256, 512, kB * 256 * 512);
  k_tout<<<ceil_div(kB * 256 * 3, 256), 256, 0, stream>>>(l3x, out3, 256, 3, kB * 256 * 3);
  {
    size_t mark = ws.off;
    float* h  = (float*)ws.take((size_t)kB * 256 * 256 * sizeof(float));
    float* o3 = (float*)ws.take((size_t)kB * 256 * 3 * sizeof(float));
    run_gemm(stream, ws, l3p, kB * 256, off1, h, 256, 0, 1);
    run_gemm(stream, ws, h, kB * 256, off2, o3, 3, 0, 0);
    k_tout<<<ceil_div(kB * 256 * 3, 256), 256, 0, stream>>>(o3, out4, 256, 3, kB * 256 * 3);
    ws.off = mark;
  }
  {
    size_t mark = ws.off;
    float* h  = (float*)ws.take((size_t)kB * 256 * 256 * sizeof(float));
    float* o1 = (float*)ws.take((size_t)kB * 256 * 1 * sizeof(float));
    run_gemm(stream, ws, l3p, kB * 256, dist1, h, 256, 0, 1);
    run_gemm(stream, ws, h, kB * 256, dist2, o1, 1, 0, 0);
    k_tout<<<ceil_div(kB * 256 * 1, 256), 256, 0, stream>>>(o1, out5, 256, 1, kB * 256 * 1);
    ws.off = mark;
  }

  (void)in_sizes; (void)n_in; (void)out_size; (void)ws_size;
}
